// Attention_51264729645450
// MI455X (gfx1250) — matrix operand order repaired and checked by reading, not compiled, not run on hardware
//
#include <hip/hip_runtime.h>

#define Bq 4
#define Sq 1024
#define NXq 1024
#define Hq 16
#define Dq 64
#define Kdim 1024
#define Mrows 4096

#if __has_builtin(__builtin_amdgcn_global_load_async_to_lds_b128)
#define HAVE_ASYNC_LDS 1
#else
#define HAVE_ASYNC_LDS 0
#endif

typedef __attribute__((ext_vector_type(16))) __bf16 v16bf;
typedef __attribute__((ext_vector_type(8))) float v8f;
typedef int v4i __attribute__((vector_size(4 * sizeof(int))));
typedef __attribute__((address_space(1))) v4i* gv4i_ptr;
typedef __attribute__((address_space(3))) v4i* lv4i_ptr;

union Frag {
    unsigned int u[8];
    v16bf v;
};

__device__ __forceinline__ unsigned short f2bf(float f) {
    union { float f; unsigned int u; } x;
    x.f = f;
    unsigned int r = x.u + 0x7FFFu + ((x.u >> 16) & 1u);  // round-to-nearest-even
    return (unsigned short)(r >> 16);
}

__device__ __forceinline__ unsigned int pack2(float a, float b) {
    return (unsigned int)f2bf(a) | ((unsigned int)f2bf(b) << 16);
}

__device__ __forceinline__ v8f wmma_bf16(const Frag& a, const Frag& b, v8f c) {
    return __builtin_amdgcn_wmma_f32_16x16x32_bf16(false, a.v, false, b.v,
                                                   (short)0, c, false, false);
}

__device__ __forceinline__ void wait_async0() {
#if HAVE_ASYNC_LDS
#if __has_builtin(__builtin_amdgcn_s_wait_asynccnt)
    __builtin_amdgcn_s_wait_asynccnt(0);
#else
    asm volatile("s_wait_asynccnt 0x0" ::: "memory");
#endif
#endif
}

// A-fragment (16x32 bf16, MxK): lanes 0-15 row M=lane, VGPR j holds K pair;
// lanes 16-31 hold the other K-octets. LDS tile is row-major, stride 32 bf16
// (= 16 dwords), so each fragment register is one 32-bit ds read.
__device__ __forceinline__ void load_a_frag(const unsigned int* w, int moff,
                                            int lane, Frag& f) {
    const int row = lane & 15, half = lane >> 4;
#pragma unroll
    for (int j = 0; j < 8; ++j) {
        const int k0 = (j < 4) ? (half * 8 + 2 * j) : (16 + half * 8 + 2 * (j - 4));
        f.u[j] = w[(moff + row) * 16 + (k0 >> 1)];
    }
}

// B-fragment (32x16 bf16, KxN): col N=lane&15, K octets as in A-fragment
// (lanes 0-15 K=0..7,16..23; lanes 16-31 K=8..15,24..31). LDS tile stored transposed [n][k],
// stride 32 bf16, so (k,k+1) pairs are one dword.
__device__ __forceinline__ void load_b_frag(const unsigned int* w, int noff,
                                            int lane, Frag& f) {
    const int n = lane & 15, khalf = lane >> 4;
#pragma unroll
    for (int j = 0; j < 8; ++j)
        f.u[j] = w[(noff + n) * 16 + ((j < 4) ? (khalf * 4 + j) : (8 + khalf * 4 + (j - 4)))];
}

// ---------------------------------------------------------------------------
// Tiled GEMM: C[M,N] = A[M,K] * W[K,N] + bias.  128x64 tile / workgroup,
// 8 waves each computing a 32x32 block (2x2 wmma frags), BK = 32.
// MODE 0: A = x (f32), epilogue scatters q/k/v (bf16 ws) + present (f32 out).
// MODE 1: A = attention output (bf16 ws; staged via GLOBAL_LOAD_ASYNC_TO_LDS
//         when available), epilogue writes f32 out + bias.
// ---------------------------------------------------------------------------
template <int MODE>
__global__ __launch_bounds__(256) void gemm_wmma(
    const float* __restrict__ Af, const unsigned short* __restrict__ Abf,
    const float* __restrict__ W, const float* __restrict__ bias, int N,
    float* __restrict__ out, float* __restrict__ present,
    unsigned short* __restrict__ qws, unsigned short* __restrict__ kws,
    unsigned short* __restrict__ vws) {
    __shared__ __align__(16) unsigned short As[128 * 32];  // 8 KB, row-major
    __shared__ __align__(16) unsigned short Bs[64 * 32];   // 4 KB, [n][k]

    const int tid = threadIdx.x;
    const int lane = tid & 31, wave = tid >> 5;
    const int wm = wave & 3, wn = wave >> 2;  // 4 (M) x 2 (N) wave grid
    const int rowbase = blockIdx.y * 128;
    const int colbase = blockIdx.x * 64;

    v8f z = {0.f, 0.f, 0.f, 0.f, 0.f, 0.f, 0.f, 0.f};
    v8f acc[2][2] = {{z, z}, {z, z}};

    for (int kb = 0; kb < Kdim; kb += 32) {
        // ---- stage A tile (128 rows x 32 k) as bf16 ----
        if (MODE == 0) {
#pragma unroll
            for (int i = 0; i < 4; ++i) {
                const int idx = tid + i * 256;
                const int r = idx >> 3, c4 = (idx & 7) * 4;
                float4 fv = *(const float4*)(Af + (size_t)(rowbase + r) * Kdim + kb + c4);
                uint2 dv;
                dv.x = pack2(fv.x, fv.y);
                dv.y = pack2(fv.z, fv.w);
                *(uint2*)(&As[r * 32 + c4]) = dv;
            }
        } else {
#if HAVE_ASYNC_LDS
            // bf16 tile rows are raw byte-copies: DMA them straight into LDS,
            // bypassing VGPRs (tracked with ASYNCcnt).
#pragma unroll
            for (int i = 0; i < 2; ++i) {
                const int idx = tid + i * 256;        // 512 x 16B chunks
                const int r = idx >> 2, c8 = (idx & 3) * 8;
                unsigned short* gsrc =
                    const_cast<unsigned short*>(Abf) + (size_t)(rowbase + r) * Kdim + kb + c8;
                __builtin_amdgcn_global_load_async_to_lds_b128(
                    (gv4i_ptr)gsrc, (lv4i_ptr)(&As[r * 32 + c8]), 0, 0);
            }
#else
#pragma unroll
            for (int i = 0; i < 4; ++i) {
                const int idx = tid + i * 256;
                const int r = idx >> 3, c4 = (idx & 7) * 4;
                uint2 dv = *(const uint2*)(Abf + (size_t)(rowbase + r) * Kdim + kb + c4);
                *(uint2*)(&As[r * 32 + c4]) = dv;
            }
#endif
        }
        // ---- stage B tile (32 k x 64 n), transposed into LDS ----
#pragma unroll
        for (int i = 0; i < 2; ++i) {
            const int idx = tid * 2 + i;
            const int k = idx >> 4, n4 = (idx & 15) * 4;
            float4 fv = *(const float4*)(W + (size_t)(kb + k) * N + colbase + n4);
            Bs[(n4 + 0) * 32 + k] = f2bf(fv.x);
            Bs[(n4 + 1) * 32 + k] = f2bf(fv.y);
            Bs[(n4 + 2) * 32 + k] = f2bf(fv.z);
            Bs[(n4 + 3) * 32 + k] = f2bf(fv.w);
        }
        // prefetch next K tile (global_prefetch_b8)
        if (kb + 32 < Kdim) {
            __builtin_prefetch(W + (size_t)(kb + 32 + (tid & 31)) * N + colbase, 0, 1);
            if (MODE == 0)
                __builtin_prefetch(Af + (size_t)(rowbase + (tid >> 1)) * Kdim + kb + 32, 0, 1);
            else
                __builtin_prefetch(Abf + (size_t)(rowbase + (tid >> 1)) * Kdim + kb + 32, 0, 1);
        }
        if (MODE == 1) wait_async0();  // drain this wave's ASYNCcnt before barrier
        __syncthreads();

        Frag a0, a1, b0, b1;
        load_a_frag((const unsigned int*)As, wm * 32, lane, a0);
        load_a_frag((const unsigned int*)As, wm * 32 + 16, lane, a1);
        load_b_frag((const unsigned int*)Bs, wn * 32, lane, b0);
        load_b_frag((const unsigned int*)Bs, wn * 32 + 16, lane, b1);

        acc[0][0] = wmma_bf16(a0, b0, acc[0][0]);
        acc[0][1] = wmma_bf16(a0, b1, acc[0][1]);
        acc[1][0] = wmma_bf16(a1, b0, acc[1][0]);
        acc[1][1] = wmma_bf16(a1, b1, acc[1][1]);
        __syncthreads();
    }

    // ---- epilogue; C layout: VGPR i -> M = i (+8 for lanes>=16), N = lane&15
    const int crow = (lane >> 4) * 8;
    const int ccol = lane & 15;
#pragma unroll
    for (int mi = 0; mi < 2; ++mi)
#pragma unroll
        for (int ni = 0; ni < 2; ++ni)
#pragma unroll
            for (int i = 0; i < 8; ++i) {
                const int row = rowbase + wm * 32 + mi * 16 + i + crow;
                const int col = colbase + wn * 32 + ni * 16 + ccol;
                const float val = acc[mi][ni][i] + bias[col];
                if (MODE == 1) {
                    out[(size_t)row * N + col] = val;
                } else {
                    const int part = col >> 10, inner = col & 1023;
                    const int h = inner >> 6, d = inner & 63;
                    const int b = row >> 10, s = row & 1023;
                    const size_t hidx = (((size_t)(b * Hq + h)) * Sq + s) * Dq + d;
                    const unsigned short bv = f2bf(val);
                    if (part == 0) {
                        qws[hidx] = bv;
                    } else {
                        if (part == 1) kws[hidx] = bv; else vws[hidx] = bv;
                        present[(size_t)(part - 1) * Bq * Hq * Sq * Dq + hidx] = val;
                    }
                }
            }
}

// ---------------------------------------------------------------------------
// Flash attention: one wave per (b,h, 16-query tile). Online softmax over
// 32-key blocks; scores & PV via v_wmma_f32_16x16x32_bf16 (f32 accumulate).
// ---------------------------------------------------------------------------
__global__ __launch_bounds__(32) void flash_attn(
    const unsigned short* __restrict__ qws, const unsigned short* __restrict__ kws,
    const unsigned short* __restrict__ vws, unsigned short* __restrict__ aws) {
    __shared__ __align__(16) unsigned short Pt[16 * 32];  // P in A layout staging
    __shared__ __align__(16) unsigned short Vt[64 * 32];  // V transposed [d][key]

    const int lane = threadIdx.x;
    const int qb = blockIdx.x * 16;
    const int bh = blockIdx.y;          // b*H + h
    const int b = bh >> 4, h = bh & 15;
    const size_t base = (size_t)bh * Sq * Dq;
    const int row = lane & 15, half = lane >> 4;

    // Q fragments for both 32-wide d-chunks, held in registers for all blocks
    Frag qf[2];
    {
        const unsigned int* qrow =
            (const unsigned int*)(qws + base + (size_t)(qb + row) * Dq);
#pragma unroll
        for (int dc = 0; dc < 2; ++dc)
#pragma unroll
            for (int j = 0; j < 8; ++j) {
                const int k0 = (j < 4) ? (half * 8 + 2 * j)
                                       : (16 + half * 8 + 2 * (j - 4));
                qf[dc].u[j] = qrow[dc * 16 + (k0 >> 1)];
            }
    }

    v8f z = {0.f, 0.f, 0.f, 0.f, 0.f, 0.f, 0.f, 0.f};
    v8f o[4] = {z, z, z, z};  // 16 q x 64 d accumulator (4 C frags)
    float m[8], l[8];
#pragma unroll
    for (int i = 0; i < 8; ++i) { m[i] = -3.0e38f; l[i] = 0.0f; }

    const int nkb = (qb + 16 + 31) >> 5;  // causal extent in 32-key blocks
    for (int kb = 0; kb < nkb; ++kb) {
        const int kbase = kb * 32;

        // stage V block transposed: Vt[d][key], 64x32 bf16
        {
            const unsigned int* vrow =
                (const unsigned int*)(vws + base + (size_t)(kbase + lane) * Dq);
#pragma unroll
            for (int w = 0; w < 32; ++w) {
                const unsigned int t = vrow[w];
                Vt[(2 * w) * 32 + lane] = (unsigned short)t;
                Vt[(2 * w + 1) * 32 + lane] = (unsigned short)(t >> 16);
            }
        }

        // scores S[16q x 32k] = Q Kt : 2 key sub-tiles x 2 d-chunks = 4 wmma.
        // k is row-major (key, d) so B-frag pairs are contiguous dwords.
        v8f s0 = z, s1 = z;
#pragma unroll
        for (int t = 0; t < 2; ++t) {
            const int key = kbase + t * 16 + row;  // B-frag col = lane&15
            const unsigned int* krow =
                (const unsigned int*)(kws + base + (size_t)key * Dq);
            Frag bf0, bf1;
#pragma unroll
            for (int j = 0; j < 8; ++j) {
                bf0.u[j] = krow[(j < 4) ? (half * 4 + j) : (8 + half * 4 + (j - 4))];
                bf1.u[j] = krow[16 + ((j < 4) ? (half * 4 + j) : (8 + half * 4 + (j - 4)))];
            }
            if (t == 0) {
                s0 = wmma_bf16(qf[0], bf0, s0);
                s0 = wmma_bf16(qf[1], bf1, s0);
            } else {
                s1 = wmma_bf16(qf[0], bf0, s1);
                s1 = wmma_bf16(qf[1], bf1, s1);
            }
        }
        __syncthreads();

        // scale, causal mask, online softmax (reductions within 16-lane halves
        // via wave32 shfl_xor butterflies), rescale O, stage P as bf16
#pragma unroll
        for (int i = 0; i < 8; ++i) {
            const int qg = qb + i + half * 8;
            float x0 = s0[i] * 0.125f;  // 1/sqrt(64)
            float x1 = s1[i] * 0.125f;
            if (kbase + row > qg) x0 = -1.0e9f;
            if (kbase + 16 + row > qg) x1 = -1.0e9f;
            float mx = fmaxf(x0, x1);
            mx = fmaxf(mx, __shfl_xor(mx, 1, 32));
            mx = fmaxf(mx, __shfl_xor(mx, 2, 32));
            mx = fmaxf(mx, __shfl_xor(mx, 4, 32));
            mx = fmaxf(mx, __shfl_xor(mx, 8, 32));
            const float mn = fmaxf(m[i], mx);
            const float sc = __expf(m[i] - mn);
            const float p0 = __expf(x0 - mn);
            const float p1 = __expf(x1 - mn);
            float rs = p0 + p1;
            rs += __shfl_xor(rs, 1, 32);
            rs += __shfl_xor(rs, 2, 32);
            rs += __shfl_xor(rs, 4, 32);
            rs += __shfl_xor(rs, 8, 32);
            l[i] = l[i] * sc + rs;
            m[i] = mn;
            o[0][i] *= sc; o[1][i] *= sc; o[2][i] *= sc; o[3][i] *= sc;
            const int prow = i + half * 8;
            Pt[prow * 32 + row] = f2bf(p0);
            Pt[prow * 32 + 16 + row] = f2bf(p1);
        }
        __syncthreads();

        // O += P[16x32] @ V[32x64] : 4 wmma across d column tiles
        Frag pf;
#pragma unroll
        for (int j = 0; j < 8; ++j) {
            const int k0 = (j < 4) ? (half * 8 + 2 * j)
                                   : (16 + half * 8 + 2 * (j - 4));
            pf.u[j] = ((const unsigned int*)Pt)[row * 16 + (k0 >> 1)];
        }
        const unsigned int* vtw = (const unsigned int*)Vt;
#pragma unroll
        for (int dt = 0; dt < 4; ++dt) {
            Frag vf;
#pragma unroll
            for (int j = 0; j < 8; ++j)
                vf.u[j] = vtw[(dt * 16 + row) * 16 + ((j < 4) ? (half * 4 + j) : (8 + half * 4 + (j - 4)))];
            o[dt] = wmma_bf16(pf, vf, o[dt]);
        }
        __syncthreads();
    }

    // epilogue: normalize by row sums, write merged-head bf16 (B,S,NX) layout
#pragma unroll
    for (int i = 0; i < 8; ++i) {
        const int srow = qb + i + half * 8;
        const float inv = 1.0f / l[i];
        const size_t rb = ((size_t)(b * Sq + srow)) * NXq + h * Dq;
#pragma unroll
        for (int dt = 0; dt < 4; ++dt)
            aws[rb + dt * 16 + row] = f2bf(o[dt][i] * inv);
    }
}

extern "C" void kernel_launch(void* const* d_in, const int* in_sizes, int n_in,
                              void* d_out, int out_size, void* d_ws, size_t ws_size,
                              hipStream_t stream) {
    (void)in_sizes; (void)n_in; (void)out_size; (void)ws_size;
    const float* x = (const float*)d_in[0];
    // d_in[1] = mask_self_attention (unused: causality computed analytically)
    const float* w_attn = (const float*)d_in[2];
    const float* b_attn = (const float*)d_in[3];
    const float* w_proj = (const float*)d_in[4];
    const float* b_proj = (const float*)d_in[5];

    float* outA = (float*)d_out;                         // (B,S,NX) f32
    float* present = outA + (size_t)Bq * Sq * NXq;       // (2,B,H,S,D) f32

    char* ws = (char*)d_ws;
    const size_t headBytes = (size_t)Bq * Hq * Sq * Dq * sizeof(unsigned short); // 8 MB
    unsigned short* qws = (unsigned short*)(ws);
    unsigned short* kws = (unsigned short*)(ws + headBytes);
    unsigned short* vws = (unsigned short*)(ws + 2 * headBytes);
    unsigned short* aws = (unsigned short*)(ws + 3 * headBytes);

    // 1) QKV projection, scatter q/k/v bf16 + present f32
    dim3 g1(3 * NXq / 64, Mrows / 128);
    gemm_wmma<0><<<g1, 256, 0, stream>>>(x, nullptr, w_attn, b_attn, 3 * NXq,
                                         nullptr, present, qws, kws, vws);

    // 2) causal flash attention per (b,h,q-tile)
    dim3 g2(Sq / 16, Bq * Hq);
    flash_attn<<<g2, 32, 0, stream>>>(qws, kws, vws, aws);

    // 3) output projection
    dim3 g3(NXq / 64, Mrows / 128);
    gemm_wmma<1><<<g3, 256, 0, stream>>>(nullptr, aws, w_proj, b_proj, NXq,
                                         outA, nullptr, nullptr, nullptr, nullptr);
}
